// _Model_67207648247829
// MI455X (gfx1250) — matrix operand order repaired and checked by reading, not compiled, not run on hardware
//
#include <hip/hip_runtime.h>
#include <hip/hip_bf16.h>
#include <math.h>

// Problem constants (match reference).
#define NN 8192
#define EE 65536
#define GG 8

typedef __attribute__((ext_vector_type(16))) _Float16 v16h;
typedef __attribute__((ext_vector_type(8)))  _Float16 v8h;
typedef __attribute__((ext_vector_type(8)))  float    v8f;
typedef __attribute__((ext_vector_type(4)))  float    v4f;

// ---------------- device helpers ----------------

__device__ inline void atomicMaxF(float* a, float v) {
  unsigned* u = (unsigned*)a;
  unsigned old = __float_as_uint(*a);
  while (__uint_as_float(old) < v) {
    unsigned assumed = old;
    old = atomicCAS(u, assumed, __float_as_uint(v));
    if (old == assumed) break;
  }
}
__device__ inline void atomicMinF(float* a, float v) {
  unsigned* u = (unsigned*)a;
  unsigned old = __float_as_uint(*a);
  while (__uint_as_float(old) > v) {
    unsigned assumed = old;
    old = atomicCAS(u, assumed, __float_as_uint(v));
    if (old == assumed) break;
  }
}

// ---------------- WMMA GEMM ----------------
// out[M,Nc] = act( A[M,K] @ W[K,Nc] + bias ), fp32 A -> f16 on the fly,
// W pre-packed into native B-fragment order, f32 accumulate.
// Tiling: block = 128 threads = 4 waves; each wave owns a 32x64 output tile
// (2 row-tiles x 4 col-tiles = 8 accumulators). Per 32-deep K step a wave
// issues 8 v_wmma against 2 A-fragment builds + 4 contiguous B loads — no LDS,
// no barriers, A reuse held in registers. act: 0=none 1=relu 2=leaky(0.01)
__global__ __launch_bounds__(128) void gemm_wmma_kernel(
    const float* __restrict__ A, int lda,
    const _Float16* __restrict__ Wp, const float* __restrict__ bias,
    float* __restrict__ out, int ldo, int K, int act) {
  const int tid  = threadIdx.x;
  const int lane = tid & 31;
  const int wave = tid >> 5;
  const int row0 = (blockIdx.x * 4 + wave) << 5;  // 32 rows per wave
  const int nt0  = blockIdx.y << 2;               // 4 column tiles (64 cols)
  const int ktiles = K >> 5;

  // A fragment addressing: lane L holds row M = L%16 of its row-tile;
  // K halves {kb..kb+7, kb+16..kb+23} with kb = (L<16) ? 0 : 8.
  const int m  = lane & 15;
  const int kb = (lane & 16) ? 8 : 0;

  v8f c00 = {}, c01 = {}, c02 = {}, c03 = {};
  v8f c10 = {}, c11 = {}, c12 = {}, c13 = {};

  const float* a0p = A + (size_t)(row0 + m) * lda + kb;
  const float* a1p = a0p + (size_t)16 * lda;
  const _Float16* b0p = Wp + ((size_t)(nt0 + 0) * ktiles * 32 + lane) * 16;
  const _Float16* b1p = Wp + ((size_t)(nt0 + 1) * ktiles * 32 + lane) * 16;
  const _Float16* b2p = Wp + ((size_t)(nt0 + 2) * ktiles * 32 + lane) * 16;
  const _Float16* b3p = Wp + ((size_t)(nt0 + 3) * ktiles * 32 + lane) * 16;

  for (int kt = 0; kt < ktiles; ++kt) {
    v4f f0 = *(const v4f*)(a0p);
    v4f f1 = *(const v4f*)(a0p + 4);
    v4f f2 = *(const v4f*)(a0p + 16);
    v4f f3 = *(const v4f*)(a0p + 20);
    v4f g0 = *(const v4f*)(a1p);
    v4f g1 = *(const v4f*)(a1p + 4);
    v4f g2 = *(const v4f*)(a1p + 16);
    v4f g3 = *(const v4f*)(a1p + 20);
    a0p += 32; a1p += 32;
    __builtin_prefetch(a0p, 0, 1);  // global_prefetch_b8 on next A tiles
    __builtin_prefetch(a1p, 0, 1);

    v16h a0, a1;
#pragma unroll
    for (int i = 0; i < 4; ++i) {
      a0[i]      = (_Float16)f0[i];
      a0[i + 4]  = (_Float16)f1[i];
      a0[i + 8]  = (_Float16)f2[i];
      a0[i + 12] = (_Float16)f3[i];
      a1[i]      = (_Float16)g0[i];
      a1[i + 4]  = (_Float16)g1[i];
      a1[i + 8]  = (_Float16)g2[i];
      a1[i + 12] = (_Float16)g3[i];
    }

    v16h b0 = *(const v16h*)b0p; b0p += 512;
    v16h b1 = *(const v16h*)b1p; b1p += 512;
    v16h b2 = *(const v16h*)b2p; b2p += 512;
    v16h b3 = *(const v16h*)b3p; b3p += 512;

    c00 = __builtin_amdgcn_wmma_f32_16x16x32_f16(false, a0, false, b0, (short)0, c00, false, false);
    c01 = __builtin_amdgcn_wmma_f32_16x16x32_f16(false, a0, false, b1, (short)0, c01, false, false);
    c02 = __builtin_amdgcn_wmma_f32_16x16x32_f16(false, a0, false, b2, (short)0, c02, false, false);
    c03 = __builtin_amdgcn_wmma_f32_16x16x32_f16(false, a0, false, b3, (short)0, c03, false, false);
    c10 = __builtin_amdgcn_wmma_f32_16x16x32_f16(false, a1, false, b0, (short)0, c10, false, false);
    c11 = __builtin_amdgcn_wmma_f32_16x16x32_f16(false, a1, false, b1, (short)0, c11, false, false);
    c12 = __builtin_amdgcn_wmma_f32_16x16x32_f16(false, a1, false, b2, (short)0, c12, false, false);
    c13 = __builtin_amdgcn_wmma_f32_16x16x32_f16(false, a1, false, b3, (short)0, c13, false, false);
  }

  // Epilogue. C layout: VGPR i, lane L -> M = i + 8*(L/16), N = L%16.
  const int m0 = (lane >> 4) << 3;
  const int nb = lane & 15;
  auto storeC = [&](const v8f& c, int rbase, int ntile) {
    const int n = (ntile << 4) + nb;
    const float bv = bias[n];
    float* orow = out + (size_t)(row0 + rbase + m0) * ldo + n;
#pragma unroll
    for (int i = 0; i < 8; ++i) {
      float v = c[i] + bv;
      if (act == 1) v = fmaxf(v, 0.f);
      else if (act == 2) v = (v > 0.f) ? v : 0.01f * v;
      orow[(size_t)i * ldo] = v;
    }
  };
  storeC(c00, 0, nt0 + 0);  storeC(c01, 0, nt0 + 1);
  storeC(c02, 0, nt0 + 2);  storeC(c03, 0, nt0 + 3);
  storeC(c10, 16, nt0 + 0); storeC(c11, 16, nt0 + 1);
  storeC(c12, 16, nt0 + 2); storeC(c13, 16, nt0 + 3);
}

// Pack W[K,Nc] (row-major fp32) into B fragments:
// packed[((nt*(K/32)+kt)*32 + lane)*16 + s] = W[kt*32 + 8*(lane/16) + (s&7) + 16*(s>>3)][nt*16 + lane%16]
__global__ void pack_w_kernel(const float* __restrict__ W,
                              _Float16* __restrict__ o, int K, int Nc) {
  int idx = blockIdx.x * 256 + threadIdx.x;
  int total = K * Nc;
  if (idx >= total) return;
  int n_in = idx & 15;
  int L    = (idx >> 4) & 31;
  int kt   = (idx >> 9) % (K >> 5);
  int nt   = idx / ((K >> 5) << 9);
  o[idx] = (_Float16)W[(size_t)(kt * 32 + ((L >> 4) << 3) + (n_in & 7) + ((n_in & 8) << 1)) * Nc + nt * 16 + (L & 15)];
}

// ---------------- elementwise / reduction kernels ----------------

__global__ void fillk(float* p, size_t n, float v) {
  size_t i = (size_t)blockIdx.x * 256 + threadIdx.x;
  if (i < n) p[i] = v;
}

// encoder layer 1: h[r,j] = relu(ea[r]*W[j] + b[j]),  j<64
__global__ void enc_first(const float* __restrict__ ea,
                          const float* __restrict__ W,
                          const float* __restrict__ b,
                          float* __restrict__ o) {
  int idx = blockIdx.x * 256 + threadIdx.x;
  if (idx >= EE * 64) return;
  int r = idx >> 6, j = idx & 63;
  float v = ea[r] * W[j] + b[j];
  o[idx] = fmaxf(v, 0.f);
}

// global-norm column stats: st[c]=sum, st[ncols+c]=sumsq
__global__ void colstat(const float* __restrict__ x, int M, int ncols,
                        float* __restrict__ st) {
  int c = blockIdx.x * blockDim.x + threadIdx.x;
  if (c >= ncols) return;
  int chunk = (M + gridDim.y - 1) / gridDim.y;
  int r0 = blockIdx.y * chunk;
  int r1 = r0 + chunk; if (r1 > M) r1 = M;
  float s = 0.f, s2 = 0.f;
  for (int r = r0; r < r1; ++r) {
    float v = x[(size_t)r * ncols + c];
    s += v; s2 += v * v;
  }
  atomicAdd(&st[c], s);
  atomicAdd(&st[ncols + c], s2);
}

// global-norm normalize (unbiased var, ddof=1)
__global__ void normk(float* __restrict__ x, int M, int ncols,
                      const float* __restrict__ st) {
  size_t idx = (size_t)blockIdx.x * 256 + threadIdx.x;
  if (idx >= (size_t)M * ncols) return;
  int c = (int)(idx % ncols);
  float s = st[c], s2 = st[ncols + c];
  float mean = s / M;
  float var = (s2 - s * s / M) / (M - 1);
  x[idx] = (x[idx] - mean) * rsqrtf(var + 1e-5f);
}

// segment stats (128-wide features) with atomics
__global__ void segstat_acc(const float* __restrict__ x,
                            const int* __restrict__ seg, int M,
                            float* __restrict__ sum, float* __restrict__ mx,
                            float* __restrict__ mn, float* __restrict__ cnt) {
  size_t idx = (size_t)blockIdx.x * 256 + threadIdx.x;
  if (idx >= (size_t)M * 128) return;
  int r = (int)(idx >> 7), c = (int)(idx & 127);
  float v = x[idx];
  int g = seg[r];
  atomicAdd(&sum[g * 128 + c], v);
  atomicMaxF(&mx[g * 128 + c], v);
  atomicMinF(&mn[g * 128 + c], v);
  if (c == 0) atomicAdd(&cnt[g], 1.f);
}

__global__ void seg_fin(const float* __restrict__ sum,
                        const float* __restrict__ mx,
                        const float* __restrict__ mn,
                        const float* __restrict__ cnt,
                        float* __restrict__ info) {
  int idx = blockIdx.x * 256 + threadIdx.x;
  if (idx >= GG * 128) return;
  int g = idx >> 7, c = idx & 127;
  float cg = fmaxf(cnt[g], 1.f);
  info[g * 384 + c]       = sum[idx] / cg;
  info[g * 384 + 128 + c] = mx[idx];
  info[g * 384 + 256 + c] = mn[idx];
}

// x[r,c] *= w[seg[r], c]
__global__ void scale_seg(float* __restrict__ x, const int* __restrict__ seg,
                          const float* __restrict__ w, int M) {
  size_t idx = (size_t)blockIdx.x * 256 + threadIdx.x;
  if (idx >= (size_t)M * 128) return;
  int r = (int)(idx >> 7), c = (int)(idx & 127);
  x[idx] *= w[(size_t)seg[r] * 128 + c];
}

// dst[E,384] = [x[row], x[col], e]   (128-wide each, float4 chunks)
__global__ void cat3_gather(const float* __restrict__ x,
                            const int* __restrict__ row,
                            const int* __restrict__ col,
                            const float* __restrict__ e,
                            float* __restrict__ dst) {
  size_t idx = (size_t)blockIdx.x * 256 + threadIdx.x;
  if (idx >= (size_t)EE * 96) return;
  int r = (int)(idx / 96), q = (int)(idx % 96);
  v4f v;
  if (q < 32)       v = ((const v4f*)(x + (size_t)row[r] * 128))[q];
  else if (q < 64)  v = ((const v4f*)(x + (size_t)col[r] * 128))[q - 32];
  else              v = ((const v4f*)(e + (size_t)r * 128))[q - 64];
  ((v4f*)(dst + (size_t)r * 384))[q] = v;
}

// dst[M,256] = [A[idxA or r]*scaleA, B[idxB or r]]   (128-wide halves)
__global__ void cat2_gather(const float* __restrict__ A,
                            const int* __restrict__ idxA,
                            const float* __restrict__ scaleA,
                            const float* __restrict__ B,
                            const int* __restrict__ idxB,
                            float* __restrict__ dst, int M) {
  size_t idx = (size_t)blockIdx.x * 256 + threadIdx.x;
  if (idx >= (size_t)M * 64) return;
  int r = (int)(idx >> 6), q = (int)(idx & 63);
  v4f v;
  if (q < 32) {
    int ra = idxA ? idxA[r] : r;
    v = ((const v4f*)(A + (size_t)ra * 128))[q];
    if (scaleA) { float s = scaleA[r]; v *= s; }
  } else {
    int rb = idxB ? idxB[r] : r;
    v = ((const v4f*)(B + (size_t)rb * 128))[q - 32];
  }
  ((v4f*)(dst + (size_t)r * 256))[q] = v;
}

// out[r] = sigmoid( dot(A[r,:K], W) + b )
__global__ void dot_sig(const float* __restrict__ A, int K,
                        const float* __restrict__ W,
                        const float* __restrict__ b,
                        float* __restrict__ out, int M) {
  int r = blockIdx.x * 256 + threadIdx.x;
  if (r >= M) return;
  const float* a = A + (size_t)r * K;
  float acc = b[0];
  for (int k = 0; k < K; ++k) acc = fmaf(a[k], W[k], acc);
  out[r] = 1.f / (1.f + __expf(-acc));
}

// scatter-mean of h[E,128] into agg[N,128] by col index
__global__ void scatter_add(const float* __restrict__ h,
                            const int* __restrict__ col,
                            float* __restrict__ agg,
                            float* __restrict__ cnt) {
  size_t idx = (size_t)blockIdx.x * 256 + threadIdx.x;
  if (idx >= (size_t)EE * 128) return;
  int r = (int)(idx >> 7), c = (int)(idx & 127);
  int nidx = col[r];
  atomicAdd(&agg[(size_t)nidx * 128 + c], h[idx]);
  if (c == 0) atomicAdd(&cnt[nidx], 1.f);
}

__global__ void divide_cnt(float* __restrict__ agg,
                           const float* __restrict__ cnt) {
  size_t idx = (size_t)blockIdx.x * 256 + threadIdx.x;
  if (idx >= (size_t)NN * 128) return;
  int r = (int)(idx >> 7);
  agg[idx] /= fmaxf(cnt[r], 1.f);
}

// tiny [G,384] -> 256 -> 128 Lin+LeakyReLU+LayerNorm MLP (one block per graph)
__global__ __launch_bounds__(256) void mlp_ln_kernel(
    const float* __restrict__ info,
    const float* __restrict__ W1, const float* __restrict__ b1,
    const float* __restrict__ W2, const float* __restrict__ b2,
    const float* __restrict__ g1, const float* __restrict__ be1,
    const float* __restrict__ g2, const float* __restrict__ be2,
    float* __restrict__ out) {
  __shared__ float sin_[384];
  __shared__ float h[256];
  __shared__ float h2[128];
  __shared__ float red[256];
  const int t = threadIdx.x;
  const int g = blockIdx.x;
  for (int i = t; i < 384; i += 256) sin_[i] = info[g * 384 + i];
  __syncthreads();

  // layer 1: 384 -> 256, leaky relu
  float acc = b1[t];
  for (int k = 0; k < 384; ++k) acc = fmaf(sin_[k], W1[k * 256 + t], acc);
  acc = (acc > 0.f) ? acc : 0.01f * acc;
  h[t] = acc;

  // LayerNorm over 256 (biased var)
  red[t] = acc; __syncthreads();
  for (int s = 128; s > 0; s >>= 1) { if (t < s) red[t] += red[t + s]; __syncthreads(); }
  float m1 = red[0] / 256.f; __syncthreads();
  float d = acc - m1;
  red[t] = d * d; __syncthreads();
  for (int s = 128; s > 0; s >>= 1) { if (t < s) red[t] += red[t + s]; __syncthreads(); }
  float v1 = red[0] / 256.f; __syncthreads();
  h[t] = d * rsqrtf(v1 + 1e-5f) * g1[t] + be1[t];
  __syncthreads();

  // layer 2: 256 -> 128, leaky relu
  float acc2 = 0.f;
  if (t < 128) {
    acc2 = b2[t];
    for (int k = 0; k < 256; ++k) acc2 = fmaf(h[k], W2[k * 128 + t], acc2);
    acc2 = (acc2 > 0.f) ? acc2 : 0.01f * acc2;
    h2[t] = acc2;
  }
  // LayerNorm over 128
  red[t] = (t < 128) ? acc2 : 0.f; __syncthreads();
  for (int s = 128; s > 0; s >>= 1) { if (t < s) red[t] += red[t + s]; __syncthreads(); }
  float m2 = red[0] / 128.f; __syncthreads();
  float d2 = (t < 128) ? (h2[t] - m2) : 0.f;
  red[t] = d2 * d2; __syncthreads();
  for (int s = 128; s > 0; s >>= 1) { if (t < s) red[t] += red[t + s]; __syncthreads(); }
  float v2 = red[0] / 128.f; __syncthreads();
  if (t < 128) out[g * 128 + t] = d2 * rsqrtf(v2 + 1e-5f) * g2[t] + be2[t];
}

__global__ void pairmean(const float* __restrict__ t, float* __restrict__ o) {
  int i = blockIdx.x * 256 + threadIdx.x;
  if (i < EE / 2) o[i] = 0.5f * (t[2 * i] + t[2 * i + 1]);
}

// ---------------- host orchestration ----------------

extern "C" void kernel_launch(void* const* d_in, const int* in_sizes, int n_in,
                              void* d_out, int out_size, void* d_ws, size_t ws_size,
                              hipStream_t stream) {
  (void)in_sizes; (void)n_in; (void)out_size; (void)ws_size;

  // ---- input mapping: edge_attr, params(pytree, dicts key-sorted), edge_index, batch_node, batch_edge
  const float* EA = (const float*)d_in[0];
  int pi = 1;
  auto NXT = [&]() { return (const float*)d_in[pi++]; };

  // params: dec_edge, enc_edge, layers, wEdgeC, wNodeC (sorted keys)
  const float *decW1 = NXT(), *decB1 = NXT(), *decW2 = NXT(), *decB2 = NXT();
  const float *encW1 = NXT(), *encB1 = NXT(), *encW2 = NXT(), *encB2 = NXT();
  struct Lin { const float *W, *b; };
  struct LayerP { Lin em1a, em1b, em2a, em2b, nm1a, nm1b, nm2a, nm2b, wAa, wAb; };
  LayerP LP[3];
  for (int l = 0; l < 3; ++l) {
    LP[l].em1a = {NXT(), NXT()}; LP[l].em1b = {NXT(), NXT()};
    LP[l].em2a = {NXT(), NXT()}; LP[l].em2b = {NXT(), NXT()};
    LP[l].nm1a = {NXT(), NXT()}; LP[l].nm1b = {NXT(), NXT()};
    LP[l].nm2a = {NXT(), NXT()}; LP[l].nm2b = {NXT(), NXT()};
    LP[l].wAa  = {NXT(), NXT()}; LP[l].wAb  = {NXT(), NXT()};
  }
  // wEdgeC: lin(W1,b1,W2,b2), ln(g1,be1,g2,be2)
  const float *wEW1 = NXT(), *wEB1 = NXT(), *wEW2 = NXT(), *wEB2 = NXT();
  const float *wEg1 = NXT(), *wEbe1 = NXT(), *wEg2 = NXT(), *wEbe2 = NXT();
  const float *wNW1 = NXT(), *wNB1 = NXT(), *wNW2 = NXT(), *wNB2 = NXT();
  const float *wNg1 = NXT(), *wNbe1 = NXT(), *wNg2 = NXT(), *wNbe2 = NXT();
  const int* edge_index = (const int*)d_in[pi++];
  const int* batch_node = (const int*)d_in[pi++];
  const int* batch_edge = (const int*)d_in[pi++];
  const int* row = edge_index;
  const int* col = edge_index + EE;

  // ---- workspace carving
  char* base = (char*)d_ws;
  auto alloc = [&](size_t bytes) -> void* {
    void* r = (void*)base;
    base += (bytes + 255) & ~(size_t)255;
    return r;
  };
  float* e    = (float*)alloc((size_t)EE * 128 * 4);
  float* x    = (float*)alloc((size_t)NN * 128 * 4);
  float* catA = (float*)alloc((size_t)EE * 384 * 4);
  float* hid  = (float*)alloc((size_t)EE * 256 * 4);
  float* out1 = (float*)alloc((size_t)EE * 128 * 4);
  float* agg  = (float*)alloc((size_t)NN * 128 * 4);
  float* wagg = (float*)alloc((size_t)EE * 4);
  float* tmpE = (float*)alloc((size_t)EE * 4);
  float* cnt  = (float*)alloc((size_t)NN * 4);
  float* st   = (float*)alloc(1024 * 4);           // column stats (<=512 cols x2)
  float* ssum = (float*)alloc(GG * 128 * 4);
  float* smax = (float*)alloc(GG * 128 * 4);
  float* smin = (float*)alloc(GG * 128 * 4);
  float* scnt = (float*)alloc(GG * 4);
  float* infoN = (float*)alloc(GG * 384 * 4);
  float* infoE = (float*)alloc(GG * 384 * 4);
  float* wNv  = (float*)alloc(GG * 128 * 4);
  float* wEv  = (float*)alloc(GG * 128 * 4);
  auto allocH = [&](int K, int Nc) { return (_Float16*)alloc((size_t)K * Nc * 2); };
  _Float16* encW2p = allocH(64, 128);
  _Float16* decW1p = allocH(128, 64);
  struct LPk { _Float16 *em1a, *em1b, *em2a, *em2b, *nm1a, *nm1b, *nm2a, *nm2b, *wAa; };
  LPk PK[3];
  for (int l = 0; l < 3; ++l) {
    PK[l].em1a = allocH(384, 256); PK[l].em1b = allocH(256, 128);
    PK[l].em2a = allocH(256, 256); PK[l].em2b = allocH(256, 128);
    PK[l].nm1a = allocH(256, 256); PK[l].nm1b = allocH(256, 128);
    PK[l].nm2a = allocH(256, 256); PK[l].nm2b = allocH(256, 128);
    PK[l].wAa  = allocH(256, 128);
  }

  // ---- launch helpers
  auto pack = [&](const float* W, _Float16* o, int K, int Nc) {
    int tot = K * Nc;
    pack_w_kernel<<<(tot + 255) / 256, 256, 0, stream>>>(W, o, K, Nc);
  };
  auto fill = [&](float* p, size_t n, float v) {
    fillk<<<(unsigned)((n + 255) / 256), 256, 0, stream>>>(p, n, v);
  };
  auto gemm = [&](const float* A, int lda, const _Float16* Wp, const float* bias,
                  float* o, int M, int K, int Nc, int act) {
    dim3 grid(M / 128, Nc / 64);  // 4 waves/block, 32 rows each
    gemm_wmma_kernel<<<grid, 128, 0, stream>>>(A, lda, Wp, bias, o, Nc, K, act);
  };
  auto gn = [&](float* buf, int M, int ncols) {
    fill(st, 2 * (size_t)ncols, 0.f);
    dim3 grid((ncols + 127) / 128, 64);
    colstat<<<grid, 128, 0, stream>>>(buf, M, ncols, st);
    size_t tot = (size_t)M * ncols;
    normk<<<(unsigned)((tot + 255) / 256), 256, 0, stream>>>(buf, M, ncols, st);
  };
  auto segstats = [&](const float* v, const int* seg, int M, float* info) {
    fill(ssum, GG * 128, 0.f);
    fill(smax, GG * 128, -INFINITY);
    fill(smin, GG * 128, INFINITY);
    fill(scnt, GG, 0.f);
    size_t tot = (size_t)M * 128;
    segstat_acc<<<(unsigned)((tot + 255) / 256), 256, 0, stream>>>(v, seg, M, ssum, smax, smin, scnt);
    seg_fin<<<(GG * 128 + 255) / 256, 256, 0, stream>>>(ssum, smax, smin, scnt, info);
  };
  auto cat2 = [&](const float* A, const int* ia, const float* sa,
                  const float* B, const int* ib, float* dst, int M) {
    size_t tot = (size_t)M * 64;
    cat2_gather<<<(unsigned)((tot + 255) / 256), 256, 0, stream>>>(A, ia, sa, B, ib, dst, M);
  };

  // ---- pack all weights (cheap, deterministic each call)
  pack(encW2, encW2p, 64, 128);
  pack(decW1, decW1p, 128, 64);
  for (int l = 0; l < 3; ++l) {
    pack(LP[l].em1a.W, PK[l].em1a, 384, 256); pack(LP[l].em1b.W, PK[l].em1b, 256, 128);
    pack(LP[l].em2a.W, PK[l].em2a, 256, 256); pack(LP[l].em2b.W, PK[l].em2b, 256, 128);
    pack(LP[l].nm1a.W, PK[l].nm1a, 256, 256); pack(LP[l].nm1b.W, PK[l].nm1b, 256, 128);
    pack(LP[l].nm2a.W, PK[l].nm2a, 256, 256); pack(LP[l].nm2b.W, PK[l].nm2b, 256, 128);
    pack(LP[l].wAa.W,  PK[l].wAa,  256, 128);
  }

  // ---- x = 0
  fill(x, (size_t)NN * 128, 0.f);

  // ---- encoder: [E,1] -> 64 (elementwise outer) -> GN -> 128 (WMMA) -> GN
  enc_first<<<(EE * 64 + 255) / 256, 256, 0, stream>>>(EA, encW1, encB1, hid);
  gn(hid, EE, 64);
  gemm(hid, 64, encW2p, encB2, e, EE, 64, 128, 1);
  gn(e, EE, 128);

  // ---- 3 message-passing layers
  for (int l = 0; l < 3; ++l) {
    // per-graph stats + tiny LN MLPs -> scaling vectors
    segstats(x, batch_node, NN, infoN);
    segstats(e, batch_edge, EE, infoE);
    mlp_ln_kernel<<<GG, 256, 0, stream>>>(infoN, wNW1, wNB1, wNW2, wNB2,
                                          wNg1, wNbe1, wNg2, wNbe2, wNv);
    mlp_ln_kernel<<<GG, 256, 0, stream>>>(infoE, wEW1, wEB1, wEW2, wEB2,
                                          wEg1, wEbe1, wEg2, wEbe2, wEv);
    scale_seg<<<(unsigned)(((size_t)NN * 128 + 255) / 256), 256, 0, stream>>>(x, batch_node, wNv, NN);
    scale_seg<<<(unsigned)(((size_t)EE * 128 + 255) / 256), 256, 0, stream>>>(e, batch_edge, wEv, EE);

    // edge_mlp1 on [xr, xc, e]
    cat3_gather<<<(unsigned)(((size_t)EE * 96 + 255) / 256), 256, 0, stream>>>(x, row, col, e, catA);
    gemm(catA, 384, PK[l].em1a, LP[l].em1a.b, hid, EE, 384, 256, 1); gn(hid, EE, 256);
    gemm(hid, 256, PK[l].em1b, LP[l].em1b.b, out1, EE, 256, 128, 1); gn(out1, EE, 128); // eo

    // edge_mlp2 on [eo, e] -> new e
    cat2(out1, nullptr, nullptr, e, nullptr, catA, EE);
    gemm(catA, 256, PK[l].em2a, LP[l].em2a.b, hid, EE, 256, 256, 1); gn(hid, EE, 256);
    gemm(hid, 256, PK[l].em2b, LP[l].em2b.b, e, EE, 256, 128, 1);    gn(e, EE, 128);

    // wAgg on [xc, xr] -> sigmoid scalar per edge
    cat2(x, col, nullptr, x, row, catA, EE);
    gemm(catA, 256, PK[l].wAa, LP[l].wAa.b, hid, EE, 256, 128, 1); gn(hid, EE, 128);
    dot_sig<<<(EE + 255) / 256, 256, 0, stream>>>(hid, 128, LP[l].wAb.W, LP[l].wAb.b, wagg, EE);

    // node_mlp1 on [xr*wAgg, e] -> h
    cat2(x, row, wagg, e, nullptr, catA, EE);
    gemm(catA, 256, PK[l].nm1a, LP[l].nm1a.b, hid, EE, 256, 256, 1); gn(hid, EE, 256);
    gemm(hid, 256, PK[l].nm1b, LP[l].nm1b.b, out1, EE, 256, 128, 1); gn(out1, EE, 128); // h

    // agg = seg_mean(h, col)
    fill(agg, (size_t)NN * 128, 0.f);
    fill(cnt, NN, 0.f);
    scatter_add<<<(unsigned)(((size_t)EE * 128 + 255) / 256), 256, 0, stream>>>(out1, col, agg, cnt);
    divide_cnt<<<(unsigned)(((size_t)NN * 128 + 255) / 256), 256, 0, stream>>>(agg, cnt);

    // node_mlp2 on [x, agg] -> new x
    cat2(x, nullptr, nullptr, agg, nullptr, catA, NN);
    gemm(catA, 256, PK[l].nm2a, LP[l].nm2a.b, hid, NN, 256, 256, 1); gn(hid, NN, 256);
    gemm(hid, 256, PK[l].nm2b, LP[l].nm2b.b, x, NN, 256, 128, 1);    gn(x, NN, 128);
  }

  // ---- decoder: 128 -> 64 (WMMA) -> GN -> 1 (dot+sigmoid) -> pair mean
  gemm(e, 128, decW1p, decB1, hid, EE, 128, 64, 1);
  gn(hid, EE, 64);
  dot_sig<<<(EE + 255) / 256, 256, 0, stream>>>(hid, 64, decW2, decB2, tmpE, EE);
  pairmean<<<(EE / 2 + 255) / 256, 256, 0, stream>>>(tmpE, (float*)d_out);
}
